// SpikingBrainGPU_7902739825143
// MI455X (gfx1250) — 8-bit matrix operand order repaired and checked by reading, compiled, not run on hardware
//
#include <hip/hip_runtime.h>
#include <cmath>
#include <cstddef>
#include <cstdint>


typedef __attribute__((ext_vector_type(8))) int v8i;

#define NN      8000
#define TSTEPS  200
#define NEXC    6400
#define KCHUNK  64
#define TREF    20
#define BLOCK   128
#define POSTS_PER_WG 64

__global__ void zero_ws_kernel(unsigned int* __restrict__ p, size_t n_dwords) {
    size_t i = (size_t)blockIdx.x * blockDim.x + threadIdx.x;
    size_t stride = (size_t)gridDim.x * blockDim.x;
    for (; i < n_dwords; i += stride) p[i] = 0u;
}

__global__ void scatter_mask_kernel(const int* __restrict__ pre,
                                    const int* __restrict__ post,
                                    unsigned char* __restrict__ mask,
                                    int n_edges) {
    int e = blockIdx.x * blockDim.x + threadIdx.x;
    if (e < n_edges) {
        mask[(size_t)post[e] * NN + pre[e]] = 1;
    }
}

__global__ __launch_bounds__(BLOCK)
void snn_step_kernel(const float* __restrict__ bg_t,
                     const float* __restrict__ tau_m,
                     const unsigned char* __restrict__ mask,
                     const unsigned char* __restrict__ sp_in,
                     unsigned char* __restrict__ sp_out,
                     float* __restrict__ v, float* __restrict__ s,
                     float* __restrict__ a, int* __restrict__ ref,
                     float* __restrict__ sum_sp, float* __restrict__ sum_v,
                     float j_eff, float j_inh, float decay_s, float decay_a)
{
    __shared__ __align__(16) unsigned char spk[NN + 16];
    {
        const unsigned int lds_base = (unsigned int)(uintptr_t)(void*)spk;
        for (int idx = threadIdx.x; idx < NN / 16; idx += BLOCK) {
            unsigned int       laddr = lds_base + (unsigned int)(idx * 16);
            unsigned long long gaddr =
                (unsigned long long)(uintptr_t)(sp_in + (size_t)idx * 16);
            asm volatile("global_load_async_to_lds_b128 %0, %1, off"
                         :: "v"(laddr), "v"(gaddr) : "memory");
        }
        asm volatile("s_wait_asynccnt 0x0" ::: "memory");
    }
    __syncthreads();

    const int wave  = threadIdx.x >> 5;
    const int lane  = threadIdx.x & 31;
    const int post0 = blockIdx.x * POSTS_PER_WG + wave * 16;
    const int col   = lane & 15;
    const int kh    = lane >> 4;

    const unsigned char* brow   = mask + (size_t)(post0 + col) * NN + kh * 16;
    const unsigned char* abase0 = spk + kh * 16;

    auto loadA = [&](int kb) -> v8i {
        const unsigned char* p = abase0 + kb;
        uint2 a0 = *(const uint2*)(p +  0);
        uint2 a1 = *(const uint2*)(p +  8);
        uint2 a2 = *(const uint2*)(p + 32);
        uint2 a3 = *(const uint2*)(p + 40);
        v8i A;
        A[0] = (int)a0.x; A[1] = (int)a0.y;
        A[2] = (int)a1.x; A[3] = (int)a1.y;
        A[4] = (int)a2.x; A[5] = (int)a2.y;
        A[6] = (int)a3.x; A[7] = (int)a3.y;
        return A;
    };
    auto loadB = [&](int kb) -> v8i {
        uint4 b0 = *(const uint4*)(brow + kb);
        uint4 b1 = *(const uint4*)(brow + kb + 32);
        v8i B;
        B[0] = (int)b0.x; B[1] = (int)b0.y; B[2] = (int)b0.z; B[3] = (int)b0.w;
        B[4] = (int)b1.x; B[5] = (int)b1.y; B[6] = (int)b1.z; B[7] = (int)b1.w;
        return B;
    };

    v8i acc_e = {};
    {
        v8i A = loadA(0);
        v8i B = loadB(0);
        for (int kb = 0; kb < NEXC - KCHUNK; kb += KCHUNK) {
            v8i An = loadA(kb + KCHUNK);
            v8i Bn = loadB(kb + KCHUNK);
            __builtin_prefetch(brow + kb + 2 * KCHUNK, 0, 1);
            acc_e = __builtin_amdgcn_wmma_i32_16x16x64_iu8(false, A, false, B,
                                                           acc_e, false, false);
            A = An; B = Bn;
        }
        acc_e = __builtin_amdgcn_wmma_i32_16x16x64_iu8(false, A, false, B,
                                                       acc_e, false, false);
    }

    v8i acc_i = {};
    {
        v8i A = loadA(NEXC);
        v8i B = loadB(NEXC);
        for (int kb = NEXC; kb < NN - KCHUNK; kb += KCHUNK) {
            v8i An = loadA(kb + KCHUNK);
            v8i Bn = loadB(kb + KCHUNK);
            __builtin_prefetch(brow + kb + 2 * KCHUNK, 0, 1);
            acc_i = __builtin_amdgcn_wmma_i32_16x16x64_iu8(false, A, false, B,
                                                           acc_i, false, false);
            A = An; B = Bn;
        }
        acc_i = __builtin_amdgcn_wmma_i32_16x16x64_iu8(false, A, false, B,
                                                       acc_i, false, false);
    }

    if (lane < 16) {
        const int i = post0 + lane;
        const float rec = j_eff * (float)acc_e[0] - j_inh * (float)acc_i[0];

        float sv = s[i], vv = v[i], av = a[i];
        int   rf = ref[i];
        const float spike_prev = (float)spk[i];

        sv = sv * decay_s + rec + bg_t[i];
        av = av * decay_a + spike_prev;

        const bool active = (rf <= 0);
        const float dv = __expf(-0.1f / tau_m[i]);
        vv = active ? (vv * dv + sv) : 0.0f;

        const float thr = 20.0f + 1.6f * av;
        const bool ns = (vv >= thr) && active;
        vv = ns ? 0.0f : vv;
        rf = ns ? TREF : (rf > 0 ? rf - 1 : 0);

        s[i] = sv; v[i] = vv; a[i] = av; ref[i] = rf;
        sp_out[i] = ns ? (unsigned char)1 : (unsigned char)0;
        sum_sp[i] += ns ? 1.0f : 0.0f;
        sum_v[i]  += vv;
    }
}

__global__ void finalize_kernel(const float* __restrict__ sum_sp,
                                const float* __restrict__ sum_v,
                                float* __restrict__ out) {
    int i = blockIdx.x * blockDim.x + threadIdx.x;
    if (i < NN) {
        out[i]      = sum_sp[i] * (1.0f / TSTEPS);
        out[NN + i] = sum_v[i]  * (1.0f / TSTEPS);
    }
}

extern "C" void kernel_launch(void* const* d_in, const int* in_sizes, int n_in,
                              void* d_out, int out_size, void* d_ws, size_t ws_size,
                              hipStream_t stream) {
    const float* bg      = (const float*)d_in[1];
    const float* tau_m   = (const float*)d_in[2];
    const int*   pre_idx = (const int*)d_in[3];
    const int*   post_idx= (const int*)d_in[4];
    const int    n_edges = in_sizes[3];
    float*       out     = (float*)d_out;

    const int   c_e     = (int)(NEXC * 0.1);
    const float j_eff   = 2.0f * (16.0f / (float)c_e);
    const float j_inh   = j_eff * 5.0f;
    const float decay_s = expf(-0.1f / 5.0f);
    const float decay_a = expf(-0.1f / 200.0f);

    unsigned char* ws = (unsigned char*)d_ws;
    size_t off = 0;
    unsigned char* mask = ws;            off += (size_t)NN * NN;
    float* v       = (float*)(ws + off); off += NN * sizeof(float);
    float* s       = (float*)(ws + off); off += NN * sizeof(float);
    float* a       = (float*)(ws + off); off += NN * sizeof(float);
    int*   ref     = (int*)  (ws + off); off += NN * sizeof(int);
    float* sum_sp  = (float*)(ws + off); off += NN * sizeof(float);
    float* sum_v   = (float*)(ws + off); off += NN * sizeof(float);
    unsigned char* sp0 = ws + off;       off += 8192;
    unsigned char* sp1 = ws + off;       off += 8192;

    const size_t n_dwords = off / 4;
    zero_ws_kernel<<<2048, 256, 0, stream>>>((unsigned int*)d_ws, n_dwords);

    scatter_mask_kernel<<<(n_edges + 255) / 256, 256, 0, stream>>>(
        pre_idx, post_idx, mask, n_edges);

    dim3 grid(NN / POSTS_PER_WG);
    for (int t = 0; t < TSTEPS; ++t) {
        const unsigned char* sp_in  = (t & 1) ? sp1 : sp0;
        unsigned char*       sp_out = (t & 1) ? sp0 : sp1;
        snn_step_kernel<<<grid, BLOCK, 0, stream>>>(
            bg + (size_t)t * NN, tau_m, mask, sp_in, sp_out,
            v, s, a, ref, sum_sp, sum_v,
            j_eff, j_inh, decay_s, decay_a);
    }

    finalize_kernel<<<(NN + 255) / 256, 256, 0, stream>>>(sum_sp, sum_v, out);
}
